// MLA_PT_23347442221516
// MI455X (gfx1250) — matrix operand order repaired and checked by reading, not compiled, not run on hardware
//
#include <hip/hip_runtime.h>
#include <hip/hip_bf16.h>
#include <math.h>

// Problem constants (from reference)
#define S_    2048
#define HID_  2048
#define NH_   16
#define QL_   1536
#define KVL_  512
#define DN_   128
#define DR_   64
#define DV_   128
#define DQ_   192      // DN + DR
#define IH_   4
#define ID_   128
#define TOPK_ 1024

typedef __attribute__((ext_vector_type(16))) __bf16 v16bf;
typedef __attribute__((ext_vector_type(8)))  float  f32x8;

// Native conversion: clang lowers to v_cvt_pk_bf16_f32 on gfx1250.
__device__ inline __bf16 f2bf(float f) { return (__bf16)f; }

__device__ inline unsigned pack2(float lo, float hi) {
  union { __bf16 b[2]; unsigned u; } p;
  p.b[0] = (__bf16)lo; p.b[1] = (__bf16)hi;
  return p.u;
}

union frag_u { uint4 q[2]; v16bf v; };

__device__ inline f32x8 zero8() {
  f32x8 z;
#pragma unroll
  for (int i = 0; i < 8; ++i) z[i] = 0.f;
  return z;
}

// Async 16-byte global -> LDS copy (per-lane), tracked by ASYNCcnt.
__device__ inline void async_copy_b128(unsigned lds_off, const void* gaddr) {
  asm volatile("global_load_async_to_lds_b128 %0, %1, off"
               :: "v"(lds_off), "v"(gaddr) : "memory");
}
__device__ inline void wait_async_le4() {
  asm volatile("s_wait_asynccnt 0x4" ::: "memory");
}
__device__ inline void wait_async_0() {
  asm volatile("s_wait_asynccnt 0x0" ::: "memory");
}

// ---------------------------------------------------------------------------
// LDS-tiled GEMM: C[M,N] = A[M,K] * W[K,N], fp32 in/out, bf16 WMMA.
// Block: 256 threads (8 waves), tile BM=128 x BN (128 or 64), BK=32.
// A tile: double-buffered raw fp32 in LDS, filled by global_load_async_to_lds_b128
//         (next K-step streams in while current computes).
// B tile: synchronous load + cvt, stored transposed [n][k] bf16 so WMMA B
//         fragments are contiguous ds_load_b128.
// Fragment layouts per CDNA5 ISA 7.12.2 (wave32):
//   A 16x32: lane row = lane&15; kb = (v<4?0:16) + (lane>=16?8:0) + 2*(v&3)
//   B 32x16: lane col = lane&15; kb = (v<4?0:16) + (lane>=16?8:0) + 2*(v&3)
//   C/D:     VGPR r -> M = r + (lane>=16?8:0), N = lane&15
// ---------------------------------------------------------------------------
template<int BN>
__launch_bounds__(256, 1)
__global__ void gemm_bf16_lds(const float* __restrict__ A, const float* __restrict__ W,
                              float* __restrict__ C, int M, int N, int K) {
  constexpr int NF = BN / 32;                 // 16-wide n-fragments per wave
  __shared__ float  Afp[2][128 * 32];         // 2 x 16 KB raw fp32 A tiles
  __shared__ __bf16 Bt[BN * 32];              // 8 or 4 KB transposed bf16 B tile
  const int tid  = threadIdx.x;
  const int lane = tid & 31, row = lane & 15, half = lane >> 4;
  const int w  = tid >> 5;
  const int wr = w & 3;                       // M subtile (32 rows)
  const int wc = w >> 2;                      // N subtile (BN/2 cols)
  const int m0 = blockIdx.y * 128;
  const int n0 = blockIdx.x * BN;
  (void)M;

  f32x8 acc[2][NF];
#pragma unroll
  for (int i = 0; i < 2; ++i)
#pragma unroll
    for (int j = 0; j < NF; ++j) acc[i][j] = zero8();

  // Per-thread slots for async A staging: thread covers 4 float4s of the tile
  const unsigned abase0 = (unsigned)(size_t)&Afp[0][0];
  const unsigned abase1 = (unsigned)(size_t)&Afp[1][0];

  // Kick off async copy of the first A tile into buffer 0
#pragma unroll
  for (int i = 0; i < 4; ++i) {
    int idx = tid + i * 256;                  // 0..1023 float4s
    int r = idx >> 3, c4 = idx & 7;
    async_copy_b128(abase0 + (unsigned)(r * 32 + c4 * 4) * 4u,
                    (const void*)(A + (size_t)(m0 + r) * K + c4 * 4));
  }

  for (int k0 = 0; k0 < K; k0 += 32) {
    const int cur = (k0 >> 5) & 1;
    const bool more = (k0 + 32) < K;
    // ---- issue async copy of NEXT A tile into the other buffer ----
    if (more) {
      unsigned nb = cur ? abase0 : abase1;
#pragma unroll
      for (int i = 0; i < 4; ++i) {
        int idx = tid + i * 256;
        int r = idx >> 3, c4 = idx & 7;
        async_copy_b128(nb + (unsigned)(r * 32 + c4 * 4) * 4u,
                        (const void*)(A + (size_t)(m0 + r) * K + k0 + 32 + c4 * 4));
      }
    }
    // ---- stage B tile (32 x BN), transposed into Bt[n][k] ----
#pragma unroll
    for (int i = 0; i < BN / 32; ++i) {
      int idx = tid + i * 256;                // 0..8*BN-1 float4s
      int r, c4;
      if (BN == 128) { r = idx >> 5; c4 = idx & 31; }
      else           { r = idx >> 4; c4 = idx & 15; }
      const float4 v = *(const float4*)(W + (size_t)(k0 + r) * N + n0 + c4 * 4);
      Bt[(c4 * 4 + 0) * 32 + r] = f2bf(v.x);
      Bt[(c4 * 4 + 1) * 32 + r] = f2bf(v.y);
      Bt[(c4 * 4 + 2) * 32 + r] = f2bf(v.z);
      Bt[(c4 * 4 + 3) * 32 + r] = f2bf(v.w);
    }
    // Current A tile's async copies (issued last iteration) must have landed;
    // only the 4 just-issued (next-tile) ops may remain outstanding.
    if (more) wait_async_le4(); else wait_async_0();
    __syncthreads();

    // ---- fragments ----
    const float* ab = &Afp[cur][0];
    v16bf af[2], bfr[NF];
#pragma unroll
    for (int i = 0; i < 2; ++i) {
      int rg = wr * 32 + i * 16 + row;
      const float4* p0 = (const float4*)(ab + rg * 32 + (half << 3));
      const float4* p1 = (const float4*)(ab + rg * 32 + 16 + (half << 3));
      float4 x0 = p0[0], x1 = p0[1], x2 = p1[0], x3 = p1[1];
      frag_u f;
      f.q[0].x = pack2(x0.x, x0.y); f.q[0].y = pack2(x0.z, x0.w);
      f.q[0].z = pack2(x1.x, x1.y); f.q[0].w = pack2(x1.z, x1.w);
      f.q[1].x = pack2(x2.x, x2.y); f.q[1].y = pack2(x2.z, x2.w);
      f.q[1].z = pack2(x3.x, x3.y); f.q[1].w = pack2(x3.z, x3.w);
      af[i] = f.v;
    }
#pragma unroll
    for (int j = 0; j < NF; ++j) {
      int cg = wc * (BN / 2) + j * 16 + row;
      frag_u f;
      f.q[0] = *(const uint4*)&Bt[cg * 32 + half * 8];
      f.q[1] = *(const uint4*)&Bt[cg * 32 + 16 + half * 8];
      bfr[j] = f.v;
    }
#pragma unroll
    for (int i = 0; i < 2; ++i)
#pragma unroll
      for (int j = 0; j < NF; ++j)
        acc[i][j] = __builtin_amdgcn_wmma_f32_16x16x32_bf16(false, af[i], false, bfr[j],
                                                            (short)0, acc[i][j], false, false);
    __syncthreads();
  }

#pragma unroll
  for (int i = 0; i < 2; ++i)
#pragma unroll
    for (int j = 0; j < NF; ++j)
#pragma unroll
      for (int r = 0; r < 8; ++r) {
        int m = m0 + wr * 32 + i * 16 + r + (half << 3);
        int n = n0 + wc * (BN / 2) + j * 16 + row;
        C[(size_t)m * N + n] = acc[i][j][r];
      }
}

// ---------------------------------------------------------------------------
// RMSNorm (row-wise), strided/in-place: y = w * x * rsqrt(mean(x^2)+eps)
// ---------------------------------------------------------------------------
__global__ void rmsnorm_kernel(const float* __restrict__ xin, const float* __restrict__ w,
                               float* __restrict__ y, int cols, int xs, int ys) {
  __shared__ float red[256];
  int row = blockIdx.x, tid = threadIdx.x;
  const float* xp = xin + (size_t)row * xs;
  float s = 0.f;
  for (int c = tid; c < cols; c += 256) { float v = xp[c]; s += v * v; }
  red[tid] = s; __syncthreads();
  for (int st = 128; st > 0; st >>= 1) { if (tid < st) red[tid] += red[tid + st]; __syncthreads(); }
  float inv = rsqrtf(red[0] / (float)cols + 1e-6f);
  float* yp = y + (size_t)row * ys;
  for (int c = tid; c < cols; c += 256) yp[c] = w[c] * xp[c] * inv;
}

// LayerNorm in-place over 128 cols (one thread per col)
__global__ void layernorm_kernel(float* __restrict__ xy, const float* __restrict__ g,
                                 const float* __restrict__ bta) {
  __shared__ float r1[128], r2[128];
  int row = blockIdx.x, tid = threadIdx.x;
  float v = xy[(size_t)row * ID_ + tid];
  r1[tid] = v; r2[tid] = v * v; __syncthreads();
  for (int st = 64; st > 0; st >>= 1) {
    if (tid < st) { r1[tid] += r1[tid + st]; r2[tid] += r2[tid + st]; }
    __syncthreads();
  }
  float mu  = r1[0] / (float)ID_;
  float var = r2[0] / (float)ID_ - mu * mu;
  float inv = rsqrtf(var + 1e-6f);
  xy[(size_t)row * ID_ + tid] = (v - mu) * inv * g[tid] + bta[tid];
}

// ---------------------------------------------------------------------------
// RoPE kernels (in-place)
// ---------------------------------------------------------------------------
__global__ void rope_q_kernel(float* __restrict__ q, const float* __restrict__ cosb,
                              const float* __restrict__ sinb) {
  int idx = blockIdx.x * blockDim.x + threadIdx.x;
  if (idx >= S_ * NH_ * 32) return;
  int i = idx & 31, h = (idx >> 5) & (NH_ - 1), s = idx >> 9;
  float c = cosb[s * 32 + i], sn = sinb[s * 32 + i];
  float* pp = q + (size_t)s * (NH_ * DQ_) + h * DQ_ + DN_ + 2 * i;
  float xr = pp[0], xi = pp[1];
  pp[0] = xr * c - xi * sn;
  pp[1] = xr * sn + xi * c;
}

__global__ void rope_kpe_kernel(float* __restrict__ kvf, const float* __restrict__ cosb,
                                const float* __restrict__ sinb) {
  int idx = blockIdx.x * blockDim.x + threadIdx.x;
  if (idx >= S_ * 32) return;
  int i = idx & 31, s = idx >> 5;
  float c = cosb[s * 32 + i], sn = sinb[s * 32 + i];
  float* pp = kvf + (size_t)s * (KVL_ + DR_) + KVL_ + 2 * i;
  float xr = pp[0], xi = pp[1];
  pp[0] = xr * c - xi * sn;
  pp[1] = xr * sn + xi * c;
}

__global__ void rope_idx_q_kernel(float* __restrict__ qi, const float* __restrict__ cosb,
                                  const float* __restrict__ sinb) {
  int idx = blockIdx.x * blockDim.x + threadIdx.x;
  if (idx >= S_ * IH_ * 32) return;
  int i = idx & 31, h = (idx >> 5) & (IH_ - 1), s = idx >> 7;
  float c = cosb[s * 32 + i], sn = sinb[s * 32 + i];
  float* pp = qi + (size_t)s * (IH_ * ID_) + h * ID_;
  float x1 = pp[i], x2 = pp[i + 32];
  pp[i]      = x1 * c - x2 * sn;
  pp[i + 32] = x1 * sn + x2 * c;
}

__global__ void rope_idx_k_kernel(float* __restrict__ ki, const float* __restrict__ cosb,
                                  const float* __restrict__ sinb) {
  int idx = blockIdx.x * blockDim.x + threadIdx.x;
  if (idx >= S_ * 32) return;
  int i = idx & 31, s = idx >> 5;
  float c = cosb[s * 32 + i], sn = sinb[s * 32 + i];
  float* pp = ki + (size_t)s * ID_;
  float x1 = pp[i], x2 = pp[i + 32];
  pp[i]      = x1 * c - x2 * sn;
  pp[i + 32] = x1 * sn + x2 * c;
}

// weights = x @ IWproj * IH^-0.5 * ID^-0.5  (N=4, plain reduction)
__global__ void iweights_kernel(const float* __restrict__ x, const float* __restrict__ P,
                                float* __restrict__ w) {
  __shared__ float red[128 * 4];
  int row = blockIdx.x, tid = threadIdx.x;
  float a0 = 0, a1 = 0, a2 = 0, a3 = 0;
  const float* xp = x + (size_t)row * HID_;
  for (int d = tid; d < HID_; d += 128) {
    float xv = xp[d];
    a0 += xv * P[d * 4 + 0]; a1 += xv * P[d * 4 + 1];
    a2 += xv * P[d * 4 + 2]; a3 += xv * P[d * 4 + 3];
  }
  red[tid * 4 + 0] = a0; red[tid * 4 + 1] = a1; red[tid * 4 + 2] = a2; red[tid * 4 + 3] = a3;
  __syncthreads();
  for (int st = 64; st > 0; st >>= 1) {
    if (tid < st) {
#pragma unroll
      for (int hh = 0; hh < 4; ++hh) red[tid * 4 + hh] += red[(tid + st) * 4 + hh];
    }
    __syncthreads();
  }
  if (tid < 4) w[(size_t)row * IH_ + tid] = red[tid] * 0.04419417382415922f; // 4^-.5 * 128^-.5
}

// ---------------------------------------------------------------------------
// Indexer score: iscore[t,s] = sum_h relu(q_idx[t,h,:].k_idx[s,:]) * w[t,h] + causal
// Two independent WMMA accumulator chains to avoid RAW hazard stalls.
// ---------------------------------------------------------------------------
__launch_bounds__(32, 1)
__global__ void indexer_score_kernel(const float* __restrict__ qidx, const float* __restrict__ kidx,
                                     const float* __restrict__ iw, float* __restrict__ iscore) {
  const int lane = threadIdx.x & 31, row = lane & 15, half = lane >> 4;
  const int t0 = blockIdx.y << 4;   // query tile
  const int s0 = blockIdx.x << 4;   // key tile
  f32x8 total = zero8();
  const float* kp = kidx + (size_t)(s0 + row) * ID_;
  for (int h = 0; h < IH_; ++h) {
    f32x8 accA = zero8(), accB = zero8();
    const float* qp = qidx + (size_t)(t0 + row) * (IH_ * ID_) + h * ID_;
#pragma unroll
    for (int kk = 0; kk < 4; ++kk) {
      int k0 = kk * 32;
      v16bf a, b;
#pragma unroll
      for (int v = 0; v < 8; ++v) {
        int kba = k0 + ((v & 3) << 1) + ((v >> 2) << 4) + (half << 3);
        float2 qv = *(const float2*)(qp + kba);
        a[2*v] = f2bf(qv.x); a[2*v+1] = f2bf(qv.y);
        int kbb = k0 + ((v & 3) << 1) + ((v >> 2) << 4) + (half << 3);
        float2 kv2 = *(const float2*)(kp + kbb);
        b[2*v] = f2bf(kv2.x); b[2*v+1] = f2bf(kv2.y);
      }
      if (kk & 1)
        accB = __builtin_amdgcn_wmma_f32_16x16x32_bf16(false, a, false, b, (short)0, accB, false, false);
      else
        accA = __builtin_amdgcn_wmma_f32_16x16x32_bf16(false, a, false, b, (short)0, accA, false, false);
    }
#pragma unroll
    for (int r = 0; r < 8; ++r) {
      float w = iw[(size_t)(t0 + r + (half << 3)) * IH_ + h];
      total[r] += fmaxf(accA[r] + accB[r], 0.f) * w;
    }
  }
#pragma unroll
  for (int r = 0; r < 8; ++r) {
    int t = t0 + r + (half << 3);
    int s = s0 + row;
    iscore[(size_t)t * S_ + s] = total[r] + ((s <= t) ? 0.f : -1.0e9f);
  }
}

// ---------------------------------------------------------------------------
// Exact per-row radix select: thresh[row] = 1024th-largest of iscore[row, :]
// ---------------------------------------------------------------------------
__global__ void radix_thresh_kernel(const float* __restrict__ iscore, float* __restrict__ thresh) {
  __shared__ unsigned hist[256];
  __shared__ unsigned sel_digit, sel_target;
  int row = blockIdx.x, tid = threadIdx.x;
  const float* rp = iscore + (size_t)row * S_;
  unsigned prefix = 0, prefmask = 0, target = TOPK_;
  for (int shift = 24; shift >= 0; shift -= 8) {
    hist[tid] = 0; __syncthreads();
    for (int c = tid; c < S_; c += 256) {
      union { float f; unsigned u; } cv; cv.f = rp[c];
      unsigned key = (cv.u & 0x80000000u) ? ~cv.u : (cv.u | 0x80000000u);
      if ((key & prefmask) == prefix) atomicAdd(&hist[(key >> shift) & 255u], 1u);
    }
    __syncthreads();
    if (tid == 0) {
      unsigned cum = 0; int d = 255;
      for (; d >= 0; --d) { cum += hist[d]; if (cum >= target) break; }
      if (d < 0) d = 0;
      sel_digit  = (unsigned)d;
      sel_target = target - (cum - hist[d]);
    }
    __syncthreads();
    prefix   |= sel_digit << shift;
    prefmask |= 255u << shift;
    target    = sel_target;
  }
  if (tid == 0) {
    union { unsigned u; float f; } out;
    out.u = (prefix & 0x80000000u) ? (prefix & 0x7FFFFFFFu) : ~prefix;
    thresh[row] = out.f;
  }
}

// ---------------------------------------------------------------------------
// Flash attention: one wave per (16-query tile, head). Streams keys in chunks
// of 32. QK^T via 6 WMMA (two independent chains), online softmax fp32,
// P re-fragmented via LDS, P.V via 8 WMMA per chunk (independent chains).
// ---------------------------------------------------------------------------
__launch_bounds__(32, 1)
__global__ void flash_attn_kernel(const float* __restrict__ q, const float* __restrict__ kvexp,
                                  const float* __restrict__ kvfull, const float* __restrict__ iscore,
                                  const float* __restrict__ thr, float* __restrict__ attn) {
  __shared__ __bf16 pLDS[16 * 32];
  const int lane = threadIdx.x & 31, row = lane & 15, half = lane >> 4;
  const int q0 = blockIdx.x << 4;
  const int h  = blockIdx.y;
  const float scale = 0.07216878364870323f;   // 192^-0.5
  const int KVSTR = NH_ * (DN_ + DV_);        // 4096

  // Preload Q A-fragments (6 K-steps of 32 over DQ=192)
  v16bf qa[6];
  const float* qp = q + (size_t)(q0 + row) * (NH_ * DQ_) + h * DQ_;
#pragma unroll
  for (int kk = 0; kk < 6; ++kk) {
#pragma unroll
    for (int v = 0; v < 8; ++v) {
      int kb = kk * 32 + ((v & 3) << 1) + ((v >> 2) << 4) + (half << 3);
      float2 qv = *(const float2*)(qp + kb);
      qa[kk][2*v] = f2bf(qv.x); qa[kk][2*v+1] = f2bf(qv.y);
    }
  }

  float th[8], mrow[8], lrow[8];
  f32x8 oacc[8];
#pragma unroll
  for (int r = 0; r < 8; ++r) {
    th[r]   = thr[q0 + r + (half << 3)];
    mrow[r] = -3.0e38f;
    lrow[r] = 0.f;
    oacc[r] = zero8();
  }

  for (int t0 = 0; t0 < S_; t0 += 32) {
    f32x8 sacc[2];
#pragma unroll
    for (int j = 0; j < 2; ++j) {
      f32x8 accA = zero8(), accB = zero8();
      int t = t0 + j * 16 + row;   // key index (B column) for this lane
      const float* kp_lo = kvexp  + (size_t)t * KVSTR + h * (DN_ + DV_);     // d in [0,128)
      const float* kp_hi = kvfull + (size_t)t * (KVL_ + DR_) + (KVL_ - DN_); // d>=128 -> rope part
#pragma unroll
      for (int kk = 0; kk < 6; ++kk) {
        const float* kp = (kk < 4) ? kp_lo : kp_hi;
        v16bf b;
#pragma unroll
        for (int v = 0; v < 8; ++v) {
          int kb = kk * 32 + ((v & 3) << 1) + ((v >> 2) << 4) + (half << 3);
          float2 kv2 = *(const float2*)(kp + kb);
          b[2*v] = f2bf(kv2.x); b[2*v+1] = f2bf(kv2.y);
        }
        if (kk & 1)
          accB = __builtin_amdgcn_wmma_f32_16x16x32_bf16(false, qa[kk], false, b, (short)0, accB, false, false);
        else
          accA = __builtin_amdgcn_wmma_f32_16x16x32_bf16(false, qa[kk], false, b, (short)0, accA, false, false);
      }
#pragma unroll
      for (int r = 0; r < 8; ++r) accA[r] += accB[r];
      sacc[j] = accA;
    }

    // Online softmax; row statistics across the 16 lanes of each wave-half
#pragma unroll
    for (int r = 0; r < 8; ++r) {
      int qg  = q0 + r + (half << 3);
      int tg0 = t0 + row, tg1 = t0 + 16 + row;
      float v0 = sacc[0][r] * scale + ((tg0 <= qg) ? 0.f : -1.0e9f)
               + ((iscore[(size_t)qg * S_ + tg0] >= th[r]) ? 0.f : -1.0e9f);
      float v1 = sacc[1][r] * scale + ((tg1 <= qg) ? 0.f : -1.0e9f)
               + ((iscore[(size_t)qg * S_ + tg1] >= th[r]) ? 0.f : -1.0e9f);
      float loc = fmaxf(v0, v1);
#pragma unroll
      for (int d = 8; d >= 1; d >>= 1) loc = fmaxf(loc, __shfl_xor(loc, d));
      float mnew = fmaxf(mrow[r], loc);
      float corr = __expf(mrow[r] - mnew);
      mrow[r] = mnew;
      float p0 = __expf(v0 - mnew);
      float p1 = __expf(v1 - mnew);
      float ps = p0 + p1;
#pragma unroll
      for (int d = 8; d >= 1; d >>= 1) ps += __shfl_xor(ps, d);
      lrow[r] = lrow[r] * corr + ps;
#pragma unroll
      for (int dt = 0; dt < 8; ++dt) oacc[dt][r] *= corr;
      pLDS[(r + (half << 3)) * 32 + row]      = f2bf(p0);
      pLDS[(r + (half << 3)) * 32 + 16 + row] = f2bf(p1);
    }
    __syncthreads();

    // Re-fragment P (16x32) as WMMA A operand (contiguous pairs in LDS)
    v16bf pa;
#pragma unroll
    for (int v = 0; v < 8; ++v) {
      int kb = ((v & 3) << 1) + ((v >> 2) << 4) + (half << 3);
      pa[2*v] = pLDS[row * 32 + kb]; pa[2*v+1] = pLDS[row * 32 + kb + 1];
    }
    // O += P . V over 8 dv tiles of 16 (independent accumulator chains)
#pragma unroll
    for (int dt = 0; dt < 8; ++dt) {
      v16bf vb;
#pragma unroll
      for (int v = 0; v < 8; ++v) {
        int kb = ((v & 3) << 1) + ((v >> 2) << 4) + (half << 3);
        const float* vp = kvexp + (size_t)(t0 + kb) * KVSTR + h * (DN_ + DV_) + DN_ + dt * 16 + row;
        vb[2*v]   = f2bf(vp[0]);
        vb[2*v+1] = f2bf(vp[KVSTR]);
      }
      oacc[dt] = __builtin_amdgcn_wmma_f32_16x16x32_bf16(false, pa, false, vb, (short)0, oacc[dt], false, false);
    }
    __syncthreads();
  }

#pragma unroll
  for (int r = 0; r < 8; ++r) {
    float invl = 1.f / lrow[r];
#pragma unroll
    for (int dt = 0; dt < 8; ++dt) {
      attn[(size_t)(q0 + r + (half << 3)) * (NH_ * DV_) + h * DV_ + dt * 16 + row] = oacc[dt][r] * invl;
    }
  }
}

// ---------------------------------------------------------------------------
// Launcher
// ---------------------------------------------------------------------------
extern "C" void kernel_launch(void* const* d_in, const int* in_sizes, int n_in,
                              void* d_out, int out_size, void* d_ws, size_t ws_size,
                              hipStream_t stream) {
  (void)in_sizes; (void)n_in; (void)out_size; (void)ws_size;
  const float* x      = (const float*)d_in[0];
  const float* cosb   = (const float*)d_in[1];
  const float* sinb   = (const float*)d_in[2];
  // d_in[3] (mask) reproduced inline (0 / -1e9 causal)
  const float* Wq_a   = (const float*)d_in[4];
  const float* q_nw   = (const float*)d_in[5];
  const float* Wq_b   = (const float*)d_in[6];
  const float* Wkv_a  = (const float*)d_in[7];
  const float* kv_nw  = (const float*)d_in[8];
  const float* Wkv_b  = (const float*)d_in[9];
  const float* Wo     = (const float*)d_in[10];
  const float* IWq_b  = (const float*)d_in[11];
  const float* IWk    = (const float*)d_in[12];
  const float* Ik_w   = (const float*)d_in[13];
  const float* Ik_b   = (const float*)d_in[14];
  const float* IWproj = (const float*)d_in[15];
  float* out = (float*)d_out;

  // Workspace layout (floats); total ~119 MB
  float* p = (float*)d_ws;
  float* qr     = p; p += (size_t)S_ * QL_;
  float* qbuf   = p; p += (size_t)S_ * NH_ * DQ_;
  float* kvfull = p; p += (size_t)S_ * (KVL_ + DR_);
  float* kv     = p; p += (size_t)S_ * KVL_;
  float* kvexp  = p; p += (size_t)S_ * NH_ * (DN_ + DV_);
  float* qidx   = p; p += (size_t)S_ * IH_ * ID_;
  float* kidx   = p; p += (size_t)S_ * ID_;
  float* iw     = p; p += (size_t)S_ * IH_;
  float* iscore = p; p += (size_t)S_ * S_;
  float* thr    = p; p += S_;
  float* attn   = p; p += (size_t)S_ * NH_ * DV_;

  dim3 b32(32), b128(128), b256(256);

#define GEMM128(Ap, Wp, Cp, Mv, Nv, Kv) \
  gemm_bf16_lds<128><<<dim3((Nv)/128, (Mv)/128), b256, 0, stream>>>(Ap, Wp, Cp, Mv, Nv, Kv)
#define GEMM64(Ap, Wp, Cp, Mv, Nv, Kv) \
  gemm_bf16_lds<64><<<dim3((Nv)/64, (Mv)/128), b256, 0, stream>>>(Ap, Wp, Cp, Mv, Nv, Kv)

  // Main path
  GEMM128(x, Wq_a, qr, S_, QL_, HID_);
  rmsnorm_kernel<<<S_, b256, 0, stream>>>(qr, q_nw, qr, QL_, QL_, QL_);
  GEMM128(qr, Wq_b, qbuf, S_, NH_ * DQ_, QL_);
  rope_q_kernel<<<(S_ * NH_ * 32 + 255) / 256, b256, 0, stream>>>(qbuf, cosb, sinb);
  GEMM64(x, Wkv_a, kvfull, S_, KVL_ + DR_, HID_);      // N=576 -> BN=64
  rope_kpe_kernel<<<(S_ * 32 + 255) / 256, b256, 0, stream>>>(kvfull, cosb, sinb);
  rmsnorm_kernel<<<S_, b256, 0, stream>>>(kvfull, kv_nw, kv, KVL_, KVL_ + DR_, KVL_);
  GEMM128(kv, Wkv_b, kvexp, S_, NH_ * (DN_ + DV_), KVL_);

  // Indexer path
  GEMM128(qr, IWq_b, qidx, S_, IH_ * ID_, QL_);
  rope_idx_q_kernel<<<(S_ * IH_ * 32 + 255) / 256, b256, 0, stream>>>(qidx, cosb, sinb);
  GEMM128(x, IWk, kidx, S_, ID_, HID_);
  layernorm_kernel<<<S_, b128, 0, stream>>>(kidx, Ik_w, Ik_b);
  rope_idx_k_kernel<<<(S_ * 32 + 255) / 256, b256, 0, stream>>>(kidx, cosb, sinb);
  iweights_kernel<<<S_, b128, 0, stream>>>(x, IWproj, iw);
  indexer_score_kernel<<<dim3(S_ / 16, S_ / 16), b32, 0, stream>>>(qidx, kidx, iw, iscore);
  radix_thresh_kernel<<<S_, b256, 0, stream>>>(iscore, thr);

  // Attention + output projection
  flash_attn_kernel<<<dim3(S_ / 16, NH_), b32, 0, stream>>>(qbuf, kvexp, kvfull, iscore, thr, attn);
  GEMM128(attn, Wo, out, S_, HID_, NH_ * DV_);

#undef GEMM128
#undef GEMM64
}
